// Pedal_26302379721334
// MI455X (gfx1250) — matrix operand order repaired and checked by reading, not compiled, not run on hardware
//
#include <hip/hip_runtime.h>
#include <hip/hip_bf16.h>
#include <stdint.h>

// Problem constants (from reference): P=4, B=256, N=65536, D=256, K=10, SCALE=10
#define P_ 4
#define B_ 256
#define N_ 65536
#define D_ 256
#define K_ 10
#define SCALE_ 10.0f

// Tiling: 128 chunks per p; each workgroup handles all 256 rows x 512 centers.
#define NC_ 128
#define CHUNK_ (N_ / NC_)        // 512 centers per chunk
#define TILE_N_ 32               // centers per TDM tile
#define T_ (CHUNK_ / TILE_N_)    // 16 tiles per chunk
#define LDSROW_ 260              // 256 dwords + 4 dword pad (TDM pad_enable)
#define SDROW_ 36                // padded f32 row stride of the transpose buffer
#define REC_ 24                  // floats per per-(p,chunk,row) partial record
#define FRAGSTRIDE_ 48           // bytes per lane-block in the f16 fragment buffer

typedef __attribute__((ext_vector_type(16))) _Float16 v16h;
typedef __attribute__((ext_vector_type(8)))  _Float16 v8h;
typedef __attribute__((ext_vector_type(4)))  _Float16 v4h;
typedef __attribute__((ext_vector_type(8)))  float    v8f;
typedef __attribute__((ext_vector_type(4)))  float    v4f;
typedef unsigned int u32;
typedef __attribute__((ext_vector_type(4))) u32 u32x4;
typedef __attribute__((ext_vector_type(8))) int i32x8;
typedef __attribute__((ext_vector_type(4))) int i32x4;

#ifndef __has_builtin
#define __has_builtin(x) 0
#endif
#if __has_builtin(__builtin_amdgcn_tensor_load_to_lds) && __has_builtin(__builtin_amdgcn_s_wait_tensorcnt)
#define USE_TDM 1
#else
#define USE_TDM 0
#endif

#if USE_TDM
// TDM DMA of a TILE_N_ x 256 f32 tile (row stride 256 f32) global -> LDS,
// padding each 1024B row by 16B so LDS row stride is LDSROW_ dwords.
// 6-arg builtin: (uint32x4 g0, int32x8 g1, int32x4, int32x4, int32x8, i32 cpol)
__device__ __forceinline__ void tdm_load_tile(const float* gsrc, u32 lds_off) {
  u32x4 g0;
  g0.x = 1u;                                   // count=1 (valid descriptor)
  g0.y = lds_off;                              // LDS byte address
  unsigned long long ga = (unsigned long long)(uintptr_t)gsrc;
  g0.z = (u32)(ga & 0xFFFFFFFFull);            // global addr lo
  g0.w = ((u32)(ga >> 32) & 0x01FFFFFFu) | (2u << 30);  // addr[56:32] | type=2
  i32x8 g1;
  // data_size=4B(code 2), pad_enable, pad_interval=7 (256 dw), pad_amount=3 (4 dw)
  g1[0] = (int)((2u << 16) | (1u << 20) | (7u << 22) | (3u << 25));
  g1[1] = (int)(256u << 16);                   // tensor_dim0 = 256 (bits 79:48 lo16)
  g1[2] = 0;                                   // dim0 hi / dim1 lo16 (65536 -> 0)
  g1[3] = (int)(1u | (256u << 16));            // dim1 hi16 = 1 (65536) | tile_dim0 = 256
  g1[4] = (int)TILE_N_;                        // tile_dim1 = 32, tile_dim2 = 0
  g1[5] = 256;                                 // tensor_dim0_stride = 256
  g1[6] = (int)(256u << 16);                   // tensor_dim1_stride lo16 = 256
  g1[7] = 0;
  i32x4 z4 = {0, 0, 0, 0};                     // 2-D tensor: groups 2/3 unused
  i32x8 z8 = {0, 0, 0, 0, 0, 0, 0, 0};
  __builtin_amdgcn_tensor_load_to_lds(g0, g1, z4, z4, z8, 0);
}
#endif

// A fragment (16x32 f16): lane<16 -> row m, f16[0..15]=K 0..15;
// lane>=16 -> K 16..31.  f32->f16 from global.
__device__ __forceinline__ v16h load_a_frag(const float* __restrict__ feature,
                                            int p, int m, int ks, int halfSel) {
  const float* fp = feature + ((size_t)p * B_ + m) * D_ + ks * 32 + halfSel * 16;
  v4f x0 = *(const v4f*)(fp + 0);
  v4f x1 = *(const v4f*)(fp + 4);
  v4f x2 = *(const v4f*)(fp + 8);
  v4f x3 = *(const v4f*)(fp + 12);
  v16h a;
#pragma unroll
  for (int i = 0; i < 4; ++i) {
    a[i]      = (_Float16)x0[i];
    a[4 + i]  = (_Float16)x1[i];
    a[8 + i]  = (_Float16)x2[i];
    a[12 + i] = (_Float16)x3[i];
  }
  return a;
}

// B fragment from the staged f16 buffer: fragment f = ks*2 + ct, lane-block of
// 32 bytes (16 f16) at stride FRAGSTRIDE_ (16B-aligned, conflict-light).
__device__ __forceinline__ v16h load_b_frag_h(const char* __restrict__ sBh,
                                              int ct, int ks, int lane) {
  const v8h* rp = (const v8h*)(sBh + (size_t)((ks * 2 + ct) * 32 + lane) * FRAGSTRIDE_);
  v8h lo = rp[0];
  v8h hi = rp[1];
  v16h b;
#pragma unroll
  for (int i = 0; i < 8; ++i) { b[i] = lo[i]; b[8 + i] = hi[i]; }
  return b;
}

__device__ __forceinline__ v8f wmma_f16(v16h a, v16h b, v8f c) {
  return __builtin_amdgcn_wmma_f32_16x16x32_f16(false, a, false, b, (short)0, c,
                                                false, false);
}

__global__ __launch_bounds__(B_) void pedal_phase1(
    const float* __restrict__ feature, const float* __restrict__ centers,
    const int* __restrict__ position, float* __restrict__ ws) {
  extern __shared__ float smem[];
  float* sB  = smem;                           // [2][TILE_N_][LDSROW_] f32 center tiles
  float* sD  = sB + 2 * TILE_N_ * LDSROW_;     // [B_][SDROW_] dot-product transpose
  float* sC2 = sD + B_ * SDROW_;               // [TILE_N_] center squared norms
  char*  sBh = (char*)(sC2 + TILE_N_);         // [16 frags][32 lanes][48B] f16 frags

  const int tid  = threadIdx.x;
  const int lane = tid & 31;
  const int wave = tid >> 5;
  const int l15  = lane & 15;
  const int lh   = lane >> 4;

  const int bid = blockIdx.x;
  const int p = bid / NC_;
  const int chunk = bid % NC_;
  const int chunkBase = chunk * CHUNK_;

  // Per-thread row state: thread t owns feature row b = t of this p.
  float f2t = 0.0f;
  {
    const float* frow = feature + ((size_t)p * B_ + tid) * D_;
#pragma unroll 8
    for (int k = 0; k < D_; k += 4) {
      v4f v = *(const v4f*)(frow + k);
      f2t += v.x * v.x + v.y * v.y + v.z * v.z + v.w * v.w;
    }
  }
  const int pos = position[tid];

  // Online logsumexp of -SCALE*d (min distance + scaled sum) and top-K state.
  float Mrun = __builtin_inff();
  float Srun = 0.0f;
  float tkd[K_];
  int   tki[K_];
#pragma unroll
  for (int q = 0; q < K_; ++q) { tkd[q] = __builtin_inff(); tki[q] = 0; }

  // Preload all A fragments (2 row-tiles x 8 k-steps) into registers (f16).
  v16h aFrag[2][8];
#pragma unroll
  for (int rt = 0; rt < 2; ++rt) {
    int m = wave * 32 + rt * 16 + l15;
#pragma unroll
    for (int ks = 0; ks < 8; ++ks) aFrag[rt][ks] = load_a_frag(feature, p, m, ks, lh);
  }

#if USE_TDM
  if (wave == 0)
    tdm_load_tile(centers + ((size_t)p * N_ + chunkBase) * D_,
                  (u32)(uintptr_t)sB);
#endif

  // Staging-pass geometry: thread handles row n = tid>>3, 8 quads of 4 dims.
  const int stRow  = tid >> 3;
  const int stQoff = tid & 7;

#pragma unroll 1
  for (int t = 0; t < T_; ++t) {
    float* bufp = sB + (t & 1) * TILE_N_ * LDSROW_;

    __syncthreads();  // (1) previous epilogue done; buffers free
#if USE_TDM
    if (wave == 0) {
      if (t + 1 < T_) {
        tdm_load_tile(centers + ((size_t)p * N_ + chunkBase + (t + 1) * TILE_N_) * D_,
                      (u32)(uintptr_t)(sB + ((t + 1) & 1) * TILE_N_ * LDSROW_));
        __builtin_amdgcn_s_wait_tensorcnt(1);  // tile t landed; t+1 in flight
      } else {
        __builtin_amdgcn_s_wait_tensorcnt(0);
      }
    }
#else
    {  // fallback: cooperative copy, 8 threads per center row
      const float* g = centers + ((size_t)p * N_ + chunkBase + t * TILE_N_) * D_
                       + (size_t)(tid >> 3) * D_ + (tid & 7) * 32;
      float* l = bufp + (tid >> 3) * LDSROW_ + (tid & 7) * 32;
#pragma unroll
      for (int i = 0; i < 8; ++i) ((v4f*)l)[i] = ((const v4f*)g)[i];
    }
#endif
    if (tid < TILE_N_) sC2[tid] = 0.0f;
    __syncthreads();  // (2) f32 tile visible, sC2 zeroed

    // Staging: convert f32 tile -> fragment-ready f16 buffer; fold in ||c||^2.
    // Element (n,k): frag = (k>>5)*2 + (n>>4); lane = (n&15) + 16*((k>>4)&1);
    // f16 index = k&15.  Quads of 4 dims keep frag/lane fixed -> one b64 store.
    {
      float c2part = 0.0f;
      const float* srcRow = bufp + (size_t)stRow * LDSROW_;
#pragma unroll
      for (int i = 0; i < 8; ++i) {
        int k = (stQoff + 8 * i) * 4;
        v4f v = *(const v4f*)(srcRow + k);
        c2part += v.x * v.x + v.y * v.y + v.z * v.z + v.w * v.w;
        v4h h;
        h[0] = (_Float16)v.x; h[1] = (_Float16)v.y;
        h[2] = (_Float16)v.z; h[3] = (_Float16)v.w;
        int frag = ((k >> 5) << 1) + (stRow >> 4);
        int flane = (stRow & 15) + (((k >> 4) & 1) << 4);
        *(v4h*)(sBh + (size_t)(frag * 32 + flane) * FRAGSTRIDE_ + (k & 15) * 2) = h;
      }
      atomicAdd(&sC2[stRow], c2part);
    }
    __syncthreads();  // (2b) f16 fragments + sC2 ready

    // GEMM: each wave computes a 32x32 tile = 2x2 WMMA tiles, K = 256 in 8 steps.
    v8f acc[2][2];
#pragma unroll
    for (int i = 0; i < 2; ++i)
#pragma unroll
      for (int j = 0; j < 2; ++j) acc[i][j] = (v8f)(0.0f);
#pragma unroll
    for (int ks = 0; ks < 8; ++ks) {
      v16h b0 = load_b_frag_h(sBh, 0, ks, lane);
      v16h b1 = load_b_frag_h(sBh, 1, ks, lane);
      acc[0][0] = wmma_f16(aFrag[0][ks], b0, acc[0][0]);
      acc[0][1] = wmma_f16(aFrag[0][ks], b1, acc[0][1]);
      acc[1][0] = wmma_f16(aFrag[1][ks], b0, acc[1][0]);
      acc[1][1] = wmma_f16(aFrag[1][ks], b1, acc[1][1]);
    }

    // Scatter accumulators to row-major transpose buffer (C layout: VGPR r ->
    // M=r (lanes 0-15) / M=r+8 (lanes 16-31), N=lane&15).
#pragma unroll
    for (int ti = 0; ti < 2; ++ti)
#pragma unroll
      for (int tj = 0; tj < 2; ++tj)
#pragma unroll
        for (int r = 0; r < 8; ++r) {
          int row = wave * 32 + ti * 16 + r + 8 * lh;
          int col = tj * 16 + l15;
          sD[(size_t)row * SDROW_ + col] = acc[ti][tj][r];
        }
    __syncthreads();  // (3) sD complete

    // Streaming epilogue: thread t scans its row's 32 new distances.
    {
      const int nbase = chunkBase + t * TILE_N_;
      const float* drow = sD + (size_t)tid * SDROW_;
#pragma unroll
      for (int j = 0; j < TILE_N_; ++j) {
        int n = nbase + j;
        if (n == pos) continue;  // masked (reference sets it to +inf)
        float dist = f2t + sC2[j] - 2.0f * drow[j];
        // branchless logsumexp update on -SCALE*dist
        float nm = fminf(Mrun, dist);
        Srun = Srun * __expf(SCALE_ * (nm - Mrun)) + __expf(SCALE_ * (nm - dist));
        Mrun = nm;
        if (dist < tkd[K_ - 1]) {  // guarded sorted insertion (ascending)
          float cd = dist; int ci = n;
#pragma unroll
          for (int q = 0; q < K_; ++q) {
            if (cd < tkd[q]) {
              float td = tkd[q]; int ti2 = tki[q];
              tkd[q] = cd; tki[q] = ci;
              cd = td; ci = ti2;
            }
          }
        }
      }
    }
  }

  // Emit per-(p,chunk,row) partial record.
  float* rec = ws + (((size_t)p * NC_ + chunk) * B_ + tid) * REC_;
  rec[0] = Mrun;
  rec[1] = Srun;
  int* reci = (int*)rec;
#pragma unroll
  for (int q = 0; q < K_; ++q) { rec[2 + q] = tkd[q]; reci[12 + q] = tki[q]; }
}

__global__ __launch_bounds__(B_) void pedal_phase2(
    const float* __restrict__ ws, const int* __restrict__ position,
    const int* __restrict__ mem_pid, float* __restrict__ out) {
  const int p = blockIdx.x;
  const int b = threadIdx.x;

  float M = __builtin_inff(), S = 0.0f;
  float tkd[K_]; int tki[K_];
#pragma unroll
  for (int q = 0; q < K_; ++q) { tkd[q] = __builtin_inff(); tki[q] = 0; }

  for (int c = 0; c < NC_; ++c) {
    const float* rec = ws + (((size_t)p * NC_ + c) * B_ + b) * REC_;
    float m2 = rec[0], s2 = rec[1];
    float nm = fminf(M, m2);
    S = S * __expf(SCALE_ * (nm - M)) + s2 * __expf(SCALE_ * (nm - m2));
    M = nm;
    const int* ri = (const int*)rec;
#pragma unroll
    for (int q = 0; q < K_; ++q) {
      float d = rec[2 + q];
      if (d < tkd[K_ - 1]) {
        float cd = d; int ci = ri[12 + q];
#pragma unroll
        for (int u = 0; u < K_; ++u) {
          if (cd < tkd[u]) {
            float td = tkd[u]; int ti = tki[u];
            tkd[u] = cd; tki[u] = ci;
            cd = td; ci = ti;
          }
        }
      }
    }
  }

  // x = logsumexp(-SCALE * top-K), y = logsumexp(-SCALE * all)
  float xs = 0.0f;
#pragma unroll
  for (int q = 0; q < K_; ++q) xs += __expf(SCALE_ * (tkd[0] - tkd[q]));
  float x = -SCALE_ * tkd[0] + __logf(xs);
  float y = -SCALE_ * M + __logf(S);
  float contrib = y - x;

  // pos_vid = mem_pid[top_idx - (top_idx > position)]
  const int pos = position[b];
  int* outi = (int*)(out + 1);
#pragma unroll
  for (int q = 0; q < K_; ++q) {
    int oi = tki[q];
    int fidx = oi - ((oi > pos) ? 1 : 0);
    outi[((size_t)p * B_ + b) * K_ + q] = mem_pid[fidx];
  }

  __shared__ float red[B_];
  red[b] = contrib;
  __syncthreads();
#pragma unroll
  for (int s = B_ / 2; s > 0; s >>= 1) {
    if (b < s) red[b] += red[b + s];
    __syncthreads();
  }
  if (b == 0) {
    float l = red[0] / (float)B_;
    if (l != l) l = 0.0f;  // NaN -> 0 per reference
    atomicAdd(out, l / (float)P_);
  }
}

extern "C" void kernel_launch(void* const* d_in, const int* in_sizes, int n_in,
                              void* d_out, int out_size, void* d_ws, size_t ws_size,
                              hipStream_t stream) {
  (void)in_sizes; (void)n_in; (void)out_size; (void)ws_size;
  const float* feature  = (const float*)d_in[0];
  const float* centers  = (const float*)d_in[1];
  const int*   position = (const int*)d_in[2];
  // d_in[3] (camid) and d_in[4] (mem_camid) are unused by the reference math.
  const int*   mem_pid  = (const int*)d_in[5];
  float* ws  = (float*)d_ws;   // needs P_*NC_*B_*REC_*4 = ~12.6 MB
  float* out = (float*)d_out;  // [0]=loss f32, [1..] pos_vid int32 (P,B,K)

  (void)hipMemsetAsync(d_out, 0, sizeof(float), stream);  // loss accumulator

  const size_t smem =
      (size_t)(2 * TILE_N_ * LDSROW_ + B_ * SDROW_ + TILE_N_) * sizeof(float)
      + (size_t)16 * 32 * FRAGSTRIDE_;  // + f16 fragment buffer
  pedal_phase1<<<dim3(P_ * NC_), dim3(B_), smem, stream>>>(feature, centers,
                                                           position, ws);
  pedal_phase2<<<dim3(P_), dim3(B_), 0, stream>>>(ws, position, mem_pid, out);
}
